// BaseBackbone_78194174591299
// MI455X (gfx1250) — matrix operand order repaired and checked by reading, not compiled, not run on hardware
//
#include <hip/hip_runtime.h>
#include <hip/hip_bf16.h>
#include <math.h>

// CDNA5 / gfx1250: wave32, WMMA 16x16x32 bf16 -> f32 accumulate.
typedef __attribute__((ext_vector_type(16))) __bf16 v16bf;
typedef __attribute__((ext_vector_type(8)))  __bf16 v8bf;
typedef __attribute__((ext_vector_type(8)))  float  v8f;

#define LN_EPS 1e-6f
static const long NT = 64L * 320 * 384;  // 7,864,320 elems of one [B,320,384] tensor

// ---------------- wave reductions (wave32) ----------------
__device__ __forceinline__ float wave_sum(float v) {
#pragma unroll
  for (int off = 16; off > 0; off >>= 1) v += __shfl_xor(v, off, 32);
  return v;
}
__device__ __forceinline__ float wave_max(float v) {
#pragma unroll
  for (int off = 16; off > 0; off >>= 1) v = fmaxf(v, __shfl_xor(v, off, 32));
  return v;
}

// ---------------- generic WMMA GEMM, bf16 x bf16 -> f32 ----------------
// One wave computes a 32x64 tile: 2 A fragments x 4 B fragments -> 8 WMMAs/k-step.
// A: bf16, K contiguous (row stride aRow elements).
// B: bf16 [N,K] layout, K contiguous (column stride bN elements).
// C: writes f32 (Cf) or bf16 (Cb), whichever pointer is non-null.
// blockIdx = (tileM32, tileN64, z); z = b1 (z%zInner, e.g. head) + b2 (z/zInner, batch).
__global__ void wmma_gemm_kernel(
    const __bf16* __restrict__ A, long aOff, long aInner, long aOuter, long aRow,
    const __bf16* __restrict__ B, long bOff, long bInner, long bOuter, long bN,
    const float* __restrict__ bias,
    const float* __restrict__ Res, long rOff, long rInner, long rOuter, long rLd,
    float* __restrict__ Cf, __bf16* __restrict__ Cb,
    long cOff, long cInner, long cOuter, long cLd,
    int K, int zInner, float alpha, int act)
{
  const int lane = threadIdx.x;                 // 0..31
  const int tm = blockIdx.x, tn = blockIdx.y;
  const int z = blockIdx.z;
  const long b1 = z % zInner, b2 = z / zInner;

  // Fragment lane mappings (ISA 7.12.2):
  // A 16x32: lanes 0-15 hold K runs {0..7},{16..23}; lanes 16-31 -> +8.
  // B 32x16: lanes 0-15 hold K runs {0..7},{16..23}; lanes 16-31 -> +8.
  const int mA   = lane & 15;
  const int nB   = lane & 15;
  const int hiKa = (lane & 16) ? 8 : 0;
  const int hiKb = (lane & 16) ? 8 : 0;

  const __bf16* aPtr0 = A + aOff + b1 * aInner + b2 * aOuter + (long)(tm * 32 + mA) * aRow;
  const __bf16* aPtr1 = aPtr0 + 16 * aRow;
  const __bf16* bPtr  = B + bOff + b1 * bInner + b2 * bOuter + (long)(tn * 64 + nB) * bN;

  v8f acc[8] = {};
  const int ksteps = K >> 5;
  for (int ks = 0; ks < ksteps; ++ks) {
    const long k0 = (long)ks << 5;

    v16bf af0, af1;
    {
      v8bf lo = *(const v8bf*)(aPtr0 + k0 + hiKa);
      v8bf hi = *(const v8bf*)(aPtr0 + k0 + 16 + hiKa);
#pragma unroll
      for (int j = 0; j < 8; ++j) { af0[j] = lo[j]; af0[j + 8] = hi[j]; }
    }
    {
      v8bf lo = *(const v8bf*)(aPtr1 + k0 + hiKa);
      v8bf hi = *(const v8bf*)(aPtr1 + k0 + 16 + hiKa);
#pragma unroll
      for (int j = 0; j < 8; ++j) { af1[j] = lo[j]; af1[j + 8] = hi[j]; }
    }

    v16bf bfr[4];
#pragma unroll
    for (int t = 0; t < 4; ++t) {
      const __bf16* bp = bPtr + (long)(t * 16) * bN + k0 + hiKb;
      v8bf lo = *(const v8bf*)(bp);
      v8bf hi = *(const v8bf*)(bp + 16);
#pragma unroll
      for (int j = 0; j < 8; ++j) { bfr[t][j] = lo[j]; bfr[t][j + 8] = hi[j]; }
    }
#pragma unroll
    for (int t = 0; t < 4; ++t)
      acc[t] = __builtin_amdgcn_wmma_f32_16x16x32_bf16(false, af0, false, bfr[t],
                                                       (short)0, acc[t], false, false);
#pragma unroll
    for (int t = 0; t < 4; ++t)
      acc[4 + t] = __builtin_amdgcn_wmma_f32_16x16x32_bf16(false, af1, false, bfr[t],
                                                           (short)0, acc[4 + t], false, false);
  }

  // C/D layout: VGPR r holds M=r (lanes 0-15) / M=r+8 (lanes 16-31), N = lane&15.
  const long cBase = cOff + b1 * cInner + b2 * cOuter;
  const long rBase = rOff + b1 * rInner + b2 * rOuter;
  const int rowHi = (lane & 16) ? 8 : 0;
#pragma unroll
  for (int half = 0; half < 2; ++half) {
#pragma unroll
    for (int t = 0; t < 4; ++t) {
      const int col = tn * 64 + t * 16 + nB;
#pragma unroll
      for (int r = 0; r < 8; ++r) {
        const int row = tm * 32 + half * 16 + r + rowHi;
        float v = acc[half * 4 + t][r] * alpha;
        if (bias) v += bias[col];
        if (act == 1) v = 0.5f * v * (1.0f + erff(v * 0.70710678118654752f));  // exact GELU
        if (Res) v += Res[rBase + (long)row * rLd + col];
        if (Cf) Cf[cBase + (long)row * cLd + col] = v;
        else    Cb[cBase + (long)row * cLd + col] = (__bf16)v;
      }
    }
  }
}

// ---------------- LayerNorm over D=384, one wave per token ----------------
__global__ void layernorm_kernel(const float* __restrict__ X, const float* __restrict__ g,
                                 const float* __restrict__ b, float* __restrict__ Yf,
                                 __bf16* __restrict__ Yb, int ntok)
{
  const int wave = threadIdx.x >> 5, lane = threadIdx.x & 31;
  const int tok = blockIdx.x * (blockDim.x >> 5) + wave;
  if (tok >= ntok) return;
  const float* x = X + (long)tok * 384;
  float s = 0.f, s2 = 0.f;
#pragma unroll
  for (int i = 0; i < 12; ++i) { float v = x[lane + i * 32]; s += v; s2 += v * v; }
  s = wave_sum(s); s2 = wave_sum(s2);
  const float mean = s * (1.f / 384.f);
  const float var  = s2 * (1.f / 384.f) - mean * mean;
  const float rstd = rsqrtf(var + LN_EPS);
#pragma unroll
  for (int i = 0; i < 12; ++i) {
    const int d = lane + i * 32;
    const float v = (x[d] - mean) * rstd * g[d] + b[d];
    if (Yf) Yf[(long)tok * 384 + d] = v;
    else    Yb[(long)tok * 384 + d] = (__bf16)v;
  }
}

// ---------------- softmax over 320 (f32 in), bf16 probs written in-place ----------------
__global__ void softmax320_kernel(float* __restrict__ S, long nrows)
{
  const int wave = threadIdx.x >> 5, lane = threadIdx.x & 31;
  const long row = (long)blockIdx.x * (blockDim.x >> 5) + wave;
  if (row >= nrows) return;
  float* p = S + row * 320;
  float vals[10]; float mx = -3.4e38f;
#pragma unroll
  for (int i = 0; i < 10; ++i) { vals[i] = p[lane + i * 32]; mx = fmaxf(mx, vals[i]); }
  mx = wave_max(mx);
  float s = 0.f;
#pragma unroll
  for (int i = 0; i < 10; ++i) { vals[i] = expf(vals[i] - mx); s += vals[i]; }
  s = wave_sum(s);
  const float inv = 1.f / s;
  // wave executes in lockstep: all loads above complete before these stores
  __bf16* pb = (__bf16*)p;   // bf16 row occupies first 640B of the 1280B f32 row
#pragma unroll
  for (int i = 0; i < 10; ++i) pb[lane + i * 32] = (__bf16)(vals[i] * inv);
}

// ---------------- weight prep: f32 [L,K,N] -> bf16 [L,N,K] ----------------
__global__ void transpose_to_bf16_kernel(const float* __restrict__ in, __bf16* __restrict__ outp,
                                         int K, int N, long total)
{
  long idx = (long)blockIdx.x * blockDim.x + threadIdx.x;
  if (idx >= total) return;
  const long KN = (long)K * N;
  const long l = idx / KN, rem = idx % KN;
  const int n = (int)(rem / K), k = (int)(rem % K);
  outp[idx] = (__bf16)in[l * KN + (long)k * N + n];
}

__global__ void cast_bf16_kernel(const float* __restrict__ in, __bf16* __restrict__ outp, long total)
{
  long idx = (long)blockIdx.x * blockDim.x + threadIdx.x;
  if (idx < total) outp[idx] = (__bf16)in[idx];
}

// ---------------- V transpose: QKVb[b,l,768+h*64+d] -> Vt[b,h,d,l] (K-contig for AV GEMM) ----
__global__ void transpose_v_kernel(const __bf16* __restrict__ QKVb, __bf16* __restrict__ Vt)
{
  const long total = 64L * 6 * 64 * 320;
  long idx = (long)blockIdx.x * blockDim.x + threadIdx.x;
  if (idx >= total) return;
  const int l = (int)(idx % 320);
  long r = idx / 320;
  const int d = (int)(r % 64); r /= 64;
  const int h = (int)(r % 6);
  const int b = (int)(r / 6);
  Vt[idx] = QKVb[(long)b * 368640 + (long)l * 1152 + 768 + h * 64 + d];
}

// ---------------- im2col for 16x16/stride16 patch embed (bf16 out) ----------------
__global__ void im2col_kernel(const float* __restrict__ z, const float* __restrict__ x,
                              __bf16* __restrict__ Ap)
{
  const long total = 64L * 320 * 768;
  long idx = (long)blockIdx.x * blockDim.x + threadIdx.x;
  if (idx >= total) return;
  const int q = (int)(idx % 768);
  long t = idx / 768;
  const int p = (int)(t % 320);
  const int b = (int)(t / 320);
  const int c = q >> 8, rem = q & 255, ii = rem >> 4, jj = rem & 15;
  float v;
  if (p < 64) {                         // z: 128x128 -> 8x8 patches
    const int pi = p >> 3, pj = p & 7;
    v = z[(((long)b * 3 + c) * 128 + pi * 16 + ii) * 128 + pj * 16 + jj];
  } else {                              // x: 256x256 -> 16x16 patches
    const int pp = p - 64, pi = pp >> 4, pj = pp & 15;
    v = x[(((long)b * 3 + c) * 256 + pi * 16 + ii) * 256 + pj * 16 + jj];
  }
  Ap[idx] = (__bf16)v;
}

// ---------------- add positional embeddings ----------------
__global__ void add_pos_kernel(float* __restrict__ T, const float* __restrict__ pz,
                               const float* __restrict__ px)
{
  long idx = (long)blockIdx.x * blockDim.x + threadIdx.x;
  if (idx >= NT) return;
  const int d = (int)(idx % 384);
  const int p = (int)((idx / 384) % 320);
  T[idx] += (p < 64) ? pz[p * 384 + d] : px[(p - 64) * 384 + d];
}

// ---------------- router MLP + top-1 select ----------------
__global__ void router_kernel(const float* __restrict__ T, const float* __restrict__ w1,
                              const float* __restrict__ b1, const float* __restrict__ w2,
                              const float* __restrict__ b2, float* __restrict__ pro,
                              int* __restrict__ sel)
{
  __shared__ float tcol[320];
  __shared__ float hid[160];
  __shared__ float p3[3];
  const int b = blockIdx.x, tid = threadIdx.x;
  for (int l = tid; l < 320; l += blockDim.x) tcol[l] = T[((long)b * 320 + l) * 384];
  __syncthreads();
  if (tid < 160) {
    float s = b1[tid];
    for (int l = 0; l < 320; ++l) s += tcol[l] * w1[l * 160 + tid];
    hid[tid] = fmaxf(s, 0.f);
  }
  __syncthreads();
  if (tid < 3) {
    float s = b2[tid];
    for (int k = 0; k < 160; ++k) s += hid[k] * w2[k * 3 + tid];
    const float pv = 1.f / (1.f + expf(-s));
    pro[b * 3 + tid] = pv;
    p3[tid] = pv;
  }
  __syncthreads();
  if (tid == 0) {
    int arg = 0; float best = p3[0];
    if (p3[1] > best) { best = p3[1]; arg = 1; }
    if (p3[2] > best) { arg = 2; }
    sel[b] = arg;                       // 0..2 (layer 9+arg)
  }
}

// ---------------- cosine similarity over token dim, mean over D ----------------
__global__ void cos_kernel(const float* __restrict__ Mid, const float* __restrict__ Tmp,
                           float* __restrict__ cosOut, int li)
{
  __shared__ float red[384];
  const int b = blockIdx.x, d = threadIdx.x;   // 384 threads
  const float* m = Mid + (long)b * 320 * 384 + d;
  const float* t = Tmp + (long)b * 320 * 384 + d;
  float smt = 0.f, smm = 0.f, stt = 0.f;
  for (int l = 0; l < 320; ++l) {
    const float mv = m[l * 384], tv = t[l * 384];
    smt += mv * tv; smm += mv * mv; stt += tv * tv;
  }
  red[d] = smt / fmaxf(sqrtf(smm) * sqrtf(stt), 1e-8f);
  __syncthreads();
  for (int st = 256; st > 0; st >>= 1) {
    if (d < st && d + st < 384) red[d] += red[d + st];
    __syncthreads();
  }
  if (d == 0) cosOut[b * 3 + li] = red[0] * (1.f / 384.f);
}

// ---------------- masked dispatch: copy selected block output ----------------
__global__ void select_copy_kernel(const float* __restrict__ Tmp, float* __restrict__ Dst,
                                   const int* __restrict__ sel, int li)
{
  long idx = (long)blockIdx.x * blockDim.x + threadIdx.x;
  if (idx >= NT) return;
  const int b = (int)(idx / (320L * 384));
  if (sel[b] == li) Dst[idx] = Tmp[idx];
}

// =================================================================
extern "C" void kernel_launch(void* const* d_in, const int* in_sizes, int n_in,
                              void* d_out, int out_size, void* d_ws, size_t ws_size,
                              hipStream_t stream) {
  const float* z      = (const float*)d_in[0];
  const float* x      = (const float*)d_in[1];
  const float* conv_w = (const float*)d_in[2];   // [384,768] flat: already [N,K]
  const float* conv_b = (const float*)d_in[3];
  const float* pos_z  = (const float*)d_in[4];
  const float* pos_x  = (const float*)d_in[5];
  const float* ln1_g  = (const float*)d_in[6];
  const float* ln1_b  = (const float*)d_in[7];
  const float* qkv_w  = (const float*)d_in[8];
  const float* qkv_b  = (const float*)d_in[9];
  const float* proj_w = (const float*)d_in[10];
  const float* proj_b = (const float*)d_in[11];
  const float* ln2_g  = (const float*)d_in[12];
  const float* ln2_b  = (const float*)d_in[13];
  const float* fc1_w  = (const float*)d_in[14];
  const float* fc1_b  = (const float*)d_in[15];
  const float* fc2_w  = (const float*)d_in[16];
  const float* fc2_b  = (const float*)d_in[17];
  const float* norm_g = (const float*)d_in[18];
  const float* norm_b = (const float*)d_in[19];
  const float* mlp1_w = (const float*)d_in[20];
  const float* mlp1_b = (const float*)d_in[21];
  const float* mlp2_w = (const float*)d_in[22];
  const float* mlp2_b = (const float*)d_in[23];

  float* out = (float*)d_out;            // [NT] out | [192] cos | [192] pro
  float* ws  = (float*)d_ws;

  // ---- workspace layout (units of f32) ----
  float*  T    = ws;                     // residual stream / mid
  float*  P    = ws + 1 * NT;            // post-attention residual
  float*  TEMP = ws + 2 * NT;            // late-block outputs
  float*  TFIN = ws + 3 * NT;            // dispatched tensor
  float*  S    = ws + 4 * NT;            // 5*NT f32: scores; overlays PAT/Fb (bf16)
  __bf16* PAT  = (__bf16*)S;             // im2col patches bf16 [20480,768]
  __bf16* Fb   = (__bf16*)S;             // fc1 output bf16 [B,320,1536]
  __bf16* QKVb = (__bf16*)(ws + 9 * NT);               // [B,320,1152] bf16 (1.5*NT)
  __bf16* Hb   = (__bf16*)(ws + 9 * NT + NT / 2 * 3);  // 10.5*NT: LN out bf16
  __bf16* Ob   = (__bf16*)(ws + 11 * NT);              // attn out bf16
  __bf16* Vt   = (__bf16*)(ws + 11 * NT + NT / 2);     // 11.5*NT: V transposed [b,h,64,320]
  long woff = 12 * NT;
  __bf16* Wqkv  = (__bf16*)(ws + woff); woff += 5308416 / 2;   // [12,1152,384]
  __bf16* Wproj = (__bf16*)(ws + woff); woff += 1769472 / 2;   // [12,384,384]
  __bf16* Wfc1  = (__bf16*)(ws + woff); woff += 7077888 / 2;   // [12,1536,384]
  __bf16* Wfc2  = (__bf16*)(ws + woff); woff += 7077888 / 2;   // [12,384,1536]
  __bf16* Wconv = (__bf16*)(ws + woff); woff += 294912 / 2;    // [384,768]
  int*    sel   = (int*)(ws + woff);

  // ---- weight prep: transpose+convert to bf16 [N,K] (K contiguous) ----
  transpose_to_bf16_kernel<<<20736, 256, 0, stream>>>(qkv_w,  Wqkv,  384,  1152, 5308416L);
  transpose_to_bf16_kernel<<< 6912, 256, 0, stream>>>(proj_w, Wproj, 384,  384,  1769472L);
  transpose_to_bf16_kernel<<<27648, 256, 0, stream>>>(fc1_w,  Wfc1,  384,  1536, 7077888L);
  transpose_to_bf16_kernel<<<27648, 256, 0, stream>>>(fc2_w,  Wfc2,  1536, 384,  7077888L);
  cast_bf16_kernel<<<1152, 256, 0, stream>>>(conv_w, Wconv, 294912L);

  // ---- patch embed: im2col(bf16) -> GEMM [20480,768]x[768,384] + conv_b -> T, +pos ----
  im2col_kernel<<<61440, 256, 0, stream>>>(z, x, PAT);
  wmma_gemm_kernel<<<dim3(640, 6, 1), 32, 0, stream>>>(
      PAT, 0, 0, 0, 768,
      Wconv, 0, 0, 0, 768,
      conv_b,
      nullptr, 0, 0, 0, 0,
      T, nullptr, 0, 0, 0, 384,
      768, 1, 1.0f, 0);
  add_pos_kernel<<<30720, 256, 0, stream>>>(T, pos_z, pos_x);

  // ---- one transformer block: input Xin, fc2 writes Cout ----
  auto run_block = [&](int i, const float* Xin, float* Cout) {
    // LN1 -> Hb (bf16)
    layernorm_kernel<<<2560, 256, 0, stream>>>(Xin, ln1_g + i * 384, ln1_b + i * 384,
                                               nullptr, Hb, 20480);
    // QKVb = Hb @ qkv_w + qkv_b (bf16 out): [320,384]x[384,1152] per batch
    wmma_gemm_kernel<<<dim3(10, 18, 64), 32, 0, stream>>>(
        Hb, 0, 0, 122880, 384,
        Wqkv + (long)i * 442368, 0, 0, 0, 384,
        qkv_b + i * 1152,
        nullptr, 0, 0, 0, 0,
        nullptr, QKVb, 0, 0, 368640, 1152,
        384, 1, 1.0f, 0);
    // Vt[b,h,d,l] = V (K-contiguous layout for the AV GEMM)
    transpose_v_kernel<<<30720, 256, 0, stream>>>(QKVb, Vt);
    // scores S = scale * Q K^T per (b,h): [320,64]x[64,320]; K rows are K-contig in QKVb
    wmma_gemm_kernel<<<dim3(10, 5, 384), 32, 0, stream>>>(
        QKVb, 0,   64, 368640, 1152,           // A = Q
        QKVb, 384, 64, 368640, 1152,           // B = K (element (k,n)=QKVb[b,n,384+h*64+k])
        nullptr, nullptr, 0, 0, 0, 0,
        S, nullptr, 0, 102400, 614400, 320,
        64, 6, 0.125f, 0);
    // softmax (f32 in, bf16 probs in-place; bf16 row stride = 640 elements)
    softmax320_kernel<<<15360, 256, 0, stream>>>(S, 122880L);
    // O = probs V per (b,h): [320,320]x[320,64] -> Ob[b, l, h*64+d]
    wmma_gemm_kernel<<<dim3(10, 1, 384), 32, 0, stream>>>(
        (const __bf16*)S, 0, 204800, 1228800, 640,   // A = probs bf16
        Vt, 0, 20480, 122880, 320,                   // B = Vt[b,h,n,k], K-contig
        nullptr, nullptr, 0, 0, 0, 0,
        nullptr, Ob, 0, 64, 122880, 384,
        320, 6, 1.0f, 0);
    // P = Xin + Ob @ proj_w + proj_b (f32 out)
    wmma_gemm_kernel<<<dim3(10, 6, 64), 32, 0, stream>>>(
        Ob, 0, 0, 122880, 384,
        Wproj + (long)i * 147456, 0, 0, 0, 384,
        proj_b + i * 384,
        Xin, 0, 0, 122880, 384,
        P, nullptr, 0, 0, 122880, 384,
        384, 1, 1.0f, 0);
    // LN2 -> Hb
    layernorm_kernel<<<2560, 256, 0, stream>>>(P, ln2_g + i * 384, ln2_b + i * 384,
                                               nullptr, Hb, 20480);
    // Fb = gelu(Hb @ fc1_w + fc1_b) (bf16 out)
    wmma_gemm_kernel<<<dim3(10, 24, 64), 32, 0, stream>>>(
        Hb, 0, 0, 122880, 384,
        Wfc1 + (long)i * 589824, 0, 0, 0, 384,
        fc1_b + i * 1536,
        nullptr, 0, 0, 0, 0,
        nullptr, Fb, 0, 0, 491520, 1536,
        384, 1, 1.0f, 1);
    // Cout = P + Fb @ fc2_w + fc2_b (f32 out)
    wmma_gemm_kernel<<<dim3(10, 6, 64), 32, 0, stream>>>(
        Fb, 0, 0, 491520, 1536,
        Wfc2 + (long)i * 589824, 0, 0, 0, 1536,
        fc2_b + i * 384,
        P, 0, 0, 122880, 384,
        Cout, nullptr, 0, 0, 122880, 384,
        1536, 1, 1.0f, 0);
  };

  // ---- dense layers 0..8 (in-place on T) ----
  for (int i = 0; i <= 8; ++i) run_block(i, T, T);

  // ---- router: pro -> d_out tail, sel -> ws ----
  router_kernel<<<64, 256, 0, stream>>>(T, mlp1_w, mlp1_b, mlp2_w, mlp2_b,
                                        out + 7864512, sel);

  // ---- late layers 9..11: temp_i = block(mid, i); cos + masked dispatch ----
  for (int i = 9; i < 12; ++i) {
    run_block(i, T, TEMP);
    cos_kernel<<<64, 384, 0, stream>>>(T, TEMP, out + 7864320, i - 9);
    select_copy_kernel<<<30720, 256, 0, stream>>>(TEMP, TFIN, sel, i - 9);
  }

  // ---- final LayerNorm -> out ----
  layernorm_kernel<<<2560, 256, 0, stream>>>(TFIN, norm_g, norm_b, out, nullptr, 20480);
}
